// HyperGraph_77077483094348
// MI455X (gfx1250) — matrix operand order repaired and checked by reading, not compiled, not run on hardware
//
#include <hip/hip_runtime.h>
#include <hip/hip_bf16.h>

typedef __attribute__((ext_vector_type(16))) _Float16 v16h;
typedef __attribute__((ext_vector_type(8)))  _Float16 v8h;
typedef __attribute__((ext_vector_type(8)))  float    v8f;
typedef __attribute__((ext_vector_type(4)))  float    v4f;

#define LDSS 40   // LDS row stride in halves (80B row pitch: 16B-aligned, conflict-spread)

// ---------------------------------------------------------------------------
// Tiled WMMA GEMM: out[M,128] = A[M,K] @ B[K,128]
//   A = H (binary incidence, f32, row length LDH=2048); TRANS_A selects H vs H^T.
//   B is pre-transposed f16: BT[128][K]  (BT[n][k] = B[k][n]).
//   A fragments are read directly from global (no redundancy: each wave owns
//   its 16 rows). B tile is double-buffered in LDS -> 1 barrier per K-step.
// ---------------------------------------------------------------------------
template<bool TRANS_A>
__global__ __launch_bounds__(256)
void gemm_h_wmma(const float* __restrict__ H,
                 const _Float16* __restrict__ BT,
                 float* __restrict__ outPart, int M, int K)
{
    constexpr int LDH = 2048;                 // E (H row length), fixed by reference
    __shared__ _Float16 Bs[2][128 * LDSS];    // 2 x 10KB

    const int tid = threadIdx.x;
    const int w   = tid >> 5;                 // wave 0..7
    const int L   = tid & 31;
    const int m0  = blockIdx.x * 128;
    const int kchunk = K / gridDim.y;
    const int k0  = blockIdx.y * kchunk;
    float* out = outPart + (size_t)blockIdx.y * (size_t)M * 128;

    v8f acc[8];
    #pragma unroll
    for (int j = 0; j < 8; ++j)
        #pragma unroll
        for (int r = 0; r < 8; ++r) acc[j][r] = 0.0f;

    // fragment addressing (ISA 7.12.2, wave32)
    const int nL  = L & 15;
    const int hiL = L >> 4;
    const int mA  = w * 16 + nL;     // A row owned by this lane
    const int kbA = hiL * 8;         // A: halves {kbA..+7, kbA+16..+23}
    const int kbB = hiL * 8;         // B: halves {kbB..+7, kbB+16..+23}

    // ---- B staging addressing: thread covers rows nS and nS+64, 8 halves each
    const int nS   = tid >> 2;                 // 0..63
    const int ksS  = (tid & 3) * 8;            // 0,8,16,24
    const int sOff = nS * LDSS + ksS;
    const _Float16* pB = BT + (size_t)nS * K + k0 + ksS;
    const size_t rowHop = (size_t)64 * K;

    // ---- A base pointer (immediate-offset loads off an advancing base)
    const float* pA = TRANS_A ? (H + (size_t)(k0 + kbA) * LDH + (m0 + mA))
                              : (H + (size_t)(m0 + mA) * LDH + (k0 + kbA));

    // prologue: stage B tile 0
    v8h rB0 = *reinterpret_cast<const v8h*>(pB);
    v8h rB1 = *reinterpret_cast<const v8h*>(pB + rowHop);
    *reinterpret_cast<v8h*>(&Bs[0][sOff])             = rB0;
    *reinterpret_cast<v8h*>(&Bs[0][64 * LDSS + sOff]) = rB1;
    pB += 32;

    int cur = 0;
    for (int kt = 0; kt < kchunk; kt += 32) {
        __syncthreads();                      // Bs[cur] ready
        const bool more = (kt + 32 < kchunk);
        if (more) {                           // fetch next B tile into regs
            rB0 = *reinterpret_cast<const v8h*>(pB);
            rB1 = *reinterpret_cast<const v8h*>(pB + rowHop);
            pB += 32;
        }

        // ---- A fragment straight from global (L2-resident H), f32 -> f16
        union { v16h v; _Float16 e[16]; } a;
        if (TRANS_A) {
            #pragma unroll
            for (int i = 0; i < 16; ++i) {
                const int krel = (i & 7) + ((i >> 3) * 16);   // {0..7,16..23}
                a.e[i] = (_Float16)pA[(size_t)krel * LDH];    // imm-offset b32 loads
            }
            pA += (size_t)32 * LDH;
        } else {
            v4f f0 = *reinterpret_cast<const v4f*>(pA);
            v4f f1 = *reinterpret_cast<const v4f*>(pA + 4);
            v4f f2 = *reinterpret_cast<const v4f*>(pA + 16);
            v4f f3 = *reinterpret_cast<const v4f*>(pA + 20);
            #pragma unroll
            for (int i = 0; i < 4; ++i) {
                a.e[i]      = (_Float16)f0[i];
                a.e[4 + i]  = (_Float16)f1[i];
                a.e[8 + i]  = (_Float16)f2[i];
                a.e[12 + i] = (_Float16)f3[i];
            }
            pA += 32;
        }
        if (more) __builtin_prefetch(pA);     // global_prefetch_b8, next A tile

        // ---- 8 WMMAs over the 16x128 strip
        #pragma unroll
        for (int j = 0; j < 8; ++j) {
            union { v16h v; v8h h[2]; } b;
            const int nB = j * 16 + nL;
            b.h[0] = *reinterpret_cast<const v8h*>(&Bs[cur][nB * LDSS + kbB]);
            b.h[1] = *reinterpret_cast<const v8h*>(&Bs[cur][nB * LDSS + kbB + 16]);
            acc[j] = __builtin_amdgcn_wmma_f32_16x16x32_f16(
                false, a.v, false, b.v, (short)0, acc[j], false, false);
        }

        if (more) {                           // stage next tile into other buffer
            *reinterpret_cast<v8h*>(&Bs[cur ^ 1][sOff])             = rB0;
            *reinterpret_cast<v8h*>(&Bs[cur ^ 1][64 * LDSS + sOff]) = rB1;
        }
        cur ^= 1;
    }

    // C/D layout: VGPR r -> (m = r + (L>>4)*8, n = L&15)
    const int mBase = w * 16 + hiL * 8;
    #pragma unroll
    for (int j = 0; j < 8; ++j)
        #pragma unroll
        for (int r = 0; r < 8; ++r)
            out[(size_t)(m0 + mBase + r) * 128 + (j * 16 + nL)] = acc[j][r];
}

// ---------------------------------------------------------------------------
// helpers
// ---------------------------------------------------------------------------
// transpose [R,128] f32 -> [128,R] f16 (B operand prep)
__global__ __launch_bounds__(256)
void transpose128_f16(const float* __restrict__ src, _Float16* __restrict__ dst, int R) {
    __shared__ float t[32][33];
    const int r0 = blockIdx.x * 32, c0 = blockIdx.y * 32;
    const int tx = threadIdx.x & 31, ty = threadIdx.x >> 5;   // ty 0..7
    #pragma unroll
    for (int yy = 0; yy < 32; yy += 8)
        t[ty + yy][tx] = src[(size_t)(r0 + ty + yy) * 128 + (c0 + tx)];
    __syncthreads();
    #pragma unroll
    for (int yy = 0; yy < 32; yy += 8)
        dst[(size_t)(c0 + ty + yy) * R + (r0 + tx)] = (_Float16)t[tx][ty + yy];
}

__global__ void deg_col(const float* __restrict__ H, float* de, int N, int E) {
    int e  = blockIdx.x * 256 + threadIdx.x;
    int n0 = blockIdx.y * 256;
    float s = 0.f;
    for (int n = n0; n < n0 + 256; ++n) s += H[(size_t)n * E + e];
    atomicAdd(&de[e], s);   // integer-valued sums -> order-invariant, deterministic
}

__global__ __launch_bounds__(256) void deg_row(const float* __restrict__ H, float* dv, int E) {
    int w = threadIdx.x >> 5, L = threadIdx.x & 31;
    int n = blockIdx.x * 8 + w;
    float s = 0.f;
    for (int e = L; e < E; e += 32) s += H[(size_t)n * E + e];
    for (int off = 16; off; off >>= 1) s += __shfl_xor(s, off, 32);
    if (L == 0) dv[n] = s;
}

__global__ void reduce_part(const float* __restrict__ part, float* dst, int len, int C) {
    int i = blockIdx.x * 256 + threadIdx.x;
    if (i < len) {
        float s = 0.f;
        for (int c = 0; c < C; ++c) s += part[(size_t)c * len + i];
        dst[i] = s;
    }
}

__global__ void scale_eh(float* eh, const float* de, const float* wE, int len) {
    int i = blockIdx.x * 256 + threadIdx.x;
    if (i < len) {
        int e = i >> 7;
        float d = de[e]; d = d < 1e-12f ? 1e-12f : d;
        eh[i] = eh[i] * (wE[e] / d);
    }
}

// node_msg = 0.5*(x + nm_raw/dv), written in place over nm_raw
__global__ void scale_nm(float* nm, const float* __restrict__ x,
                         const float* __restrict__ dv, int len) {
    int i = blockIdx.x * 256 + threadIdx.x;
    if (i < len) {
        int n = i >> 7;
        float d = dv[n]; d = d < 1e-12f ? 1e-12f : d;
        nm[i] = 0.5f * (x[i] + nm[i] / d);
    }
}

__global__ void scale_ens(float* ens, const float* de, int len) {
    int i = blockIdx.x * 256 + threadIdx.x;
    if (i < len) {
        int e = i >> 7;
        float c = de[e]; c = c < 1.0f ? 1.0f : c;   // cnt >= 1 by construction
        ens[i] = ens[i] / c;
    }
}

// wave-per-edge: scores[e] = tanh([ens|eh] @ Wk + bk) . q
__global__ __launch_bounds__(256)
void attn_scores(const float* __restrict__ ens, const float* __restrict__ eh,
                 const float* __restrict__ Wk, const float* __restrict__ bk,
                 const float* __restrict__ q, float* scores, int E) {
    int w = threadIdx.x >> 5, j = threadIdx.x & 31;
    int e = blockIdx.x * 8 + w;
    if (e >= E) return;
    float s = bk[j];
    const float* r0 = ens + (size_t)e * 128;
    const float* r1 = eh  + (size_t)e * 128;
    for (int i = 0; i < 128; ++i) s += r0[i] * Wk[i * 32 + j];
    for (int i = 0; i < 128; ++i) s += r1[i] * Wk[(128 + i) * 32 + j];
    float t = tanhf(s) * q[j];
    for (int off = 16; off; off >>= 1) t += __shfl_xor(t, off, 32);
    if (j == 0) scores[e] = t;
}

// single block: softmax over E scores, then out[256] = alpha @ [ens|eh]
__global__ __launch_bounds__(256)
void attn_finalize(const float* __restrict__ scores, const float* __restrict__ ens,
                   const float* __restrict__ eh, float* out, int E) {
    __shared__ float red[256];
    int t = threadIdx.x;
    float m = -1e30f;
    for (int e = t; e < E; e += 256) m = fmaxf(m, scores[e]);
    red[t] = m; __syncthreads();
    for (int s = 128; s; s >>= 1) { if (t < s) red[t] = fmaxf(red[t], red[t + s]); __syncthreads(); }
    float mx = red[0]; __syncthreads();
    float z = 0.f;
    for (int e = t; e < E; e += 256) z += expf(scores[e] - mx);
    red[t] = z; __syncthreads();
    for (int s = 128; s; s >>= 1) { if (t < s) red[t] += red[t + s]; __syncthreads(); }
    float Z = red[0];
    const float* src = (t < 128) ? (ens + t) : (eh + (t - 128));
    float o = 0.f;
    for (int e = 0; e < E; ++e) o += expf(scores[e] - mx) * src[(size_t)e * 128];
    out[t] = o / Z;
}

// ---------------------------------------------------------------------------
extern "C" void kernel_launch(void* const* d_in, const int* in_sizes, int n_in,
                              void* d_out, int out_size, void* d_ws, size_t ws_size,
                              hipStream_t stream) {
    const float* x  = (const float*)d_in[0];  // [N,128]
    const float* H  = (const float*)d_in[1];  // [N,E]
    const float* wE = (const float*)d_in[2];  // [E]
    const float* Wk = (const float*)d_in[3];  // [256,32]
    const float* bk = (const float*)d_in[4];  // [32]
    const float* q  = (const float*)d_in[5];  // [32]
    float* out = (float*)d_out;

    const int N = in_sizes[0] / 128;          // 16384
    const int E = in_sizes[2];                // 2048
    const int KSPLIT = 8;                     // for H^T GEMMs (K = N)

    // workspace layout (256B aligned)
    char* p = (char*)d_ws;
    auto take = [&](size_t bytes) { void* r = (void*)p; p += (bytes + 255) & ~(size_t)255; return r; };
    float*    de     = (float*)   take((size_t)E * 4);
    float*    dv     = (float*)   take((size_t)N * 4);
    float*    eh     = (float*)   take((size_t)E * 128 * 4);   // edge_hid (raw -> scaled in place)
    float*    ens    = (float*)   take((size_t)E * 128 * 4);   // edge_node_mean
    float*    nm     = (float*)   take((size_t)N * 128 * 4);   // node_msg (raw -> scaled in place)
    _Float16* xT     = (_Float16*)take((size_t)N * 128 * 2);   // x^T   [128,N]
    _Float16* ehT    = (_Float16*)take((size_t)E * 128 * 2);   // eh^T  [128,E]
    _Float16* nmT    = (_Float16*)take((size_t)N * 128 * 2);   // nm^T  [128,N]
    float*    part   = (float*)   take((size_t)KSPLIT * E * 128 * 4);
    float*    scores = (float*)   take((size_t)E * 4);

    hipMemsetAsync(de, 0, (size_t)E * 4, stream);

    // degrees + transposed f16 copy of x
    deg_col<<<dim3(E / 256, N / 256), 256, 0, stream>>>(H, de, N, E);
    deg_row<<<N / 8, 256, 0, stream>>>(H, dv, E);
    transpose128_f16<<<dim3(N / 32, 4), 256, 0, stream>>>(x, xT, N);

    // GEMM1: eh_raw = H^T @ x   (M=E, K=N, deterministic split-K)
    gemm_h_wmma<true><<<dim3(E / 128, KSPLIT), 256, 0, stream>>>(H, xT, part, E, N);
    reduce_part<<<(E * 128) / 256, 256, 0, stream>>>(part, eh, E * 128, KSPLIT);
    scale_eh<<<(E * 128) / 256, 256, 0, stream>>>(eh, de, wE, E * 128);
    transpose128_f16<<<dim3(E / 32, 4), 256, 0, stream>>>(eh, ehT, E);

    // GEMM2: nm_raw = H @ edge_hid   (M=N, K=E, no split)
    gemm_h_wmma<false><<<dim3(N / 128, 1), 256, 0, stream>>>(H, ehT, nm, N, E);
    scale_nm<<<(N * 128) / 256, 256, 0, stream>>>(nm, x, dv, N * 128);
    transpose128_f16<<<dim3(N / 32, 4), 256, 0, stream>>>(nm, nmT, N);

    // GEMM3: ens_raw = H^T @ node_msg
    gemm_h_wmma<true><<<dim3(E / 128, KSPLIT), 256, 0, stream>>>(H, nmT, part, E, N);
    reduce_part<<<(E * 128) / 256, 256, 0, stream>>>(part, ens, E * 128, KSPLIT);
    scale_ens<<<(E * 128) / 256, 256, 0, stream>>>(ens, de, E * 128);

    // attention pooling
    attn_scores<<<E / 8, 256, 0, stream>>>(ens, eh, Wk, bk, q, scores, E);
    attn_finalize<<<1, 256, 0, stream>>>(scores, ens, eh, out, E);
}
